// TwoReacModel_45655502356632
// MI455X (gfx1250) — matrix operand order repaired and checked by reading, not compiled, not run on hardware
//
#include <hip/hip_runtime.h>
#include <cstdint>

typedef __attribute__((ext_vector_type(16))) _Float16 v16h;
typedef __attribute__((ext_vector_type(8)))  float    v8f;

namespace {

constexpr int   kT      = 512;
constexpr int   kB      = 2048;
constexpr float kDelta  = 0.1f;
constexpr float kInvTau = 0.2f;   // 1/TAU, TAU = 5

union Frag {
  v16h     h;
  uint32_t u[8];
};

__device__ __forceinline__ uint32_t pk2(float a, float b) {
  // v_cvt_pk_rtz_f16_f32 -> packed half2 as u32
  return __builtin_bit_cast(uint32_t, __builtin_amdgcn_cvt_pkrtz(a, b));
}

__device__ __forceinline__ uint32_t swapx16_u32(uint32_t v) {
  // ds_swizzle_b32 group-of-32: and=0x1f, or=0, xor=0x10 -> exchange lane <-> lane^16
  return (uint32_t)__builtin_amdgcn_ds_swizzle((int)v, 0x401F);
}

__device__ __forceinline__ float swapx16_f32(float v) {
  return __builtin_bit_cast(float, swapx16_u32(__builtin_bit_cast(uint32_t, v)));
}

__device__ __forceinline__ float fast_tanh(float x) {
#if __has_builtin(__builtin_amdgcn_tanhf)
  return __builtin_amdgcn_tanhf(x);            // v_tanh_f32 (CDNA5 trans op)
#else
  // tanh(x) = 1 - 2/(exp2(2*log2(e)*x) + 1); v_exp_f32 + v_rcp_f32
  float e = __builtin_amdgcn_exp2f(2.88539008177792681f * x);
  return 1.0f - 2.0f * __builtin_amdgcn_rcpf(e + 1.0f);
#endif
}

} // namespace

// One wave (32 lanes) integrates a tile of 16 trajectories over all 512 steps.
// Activations are kept transposed: D = W^T x h^T, weights live in A-fragments
// in VGPRs for the whole kernel; batch column = lane % 16 (replicated in both
// half-waves).
extern "C" __global__ __launch_bounds__(32)
void tworeac_rk4_wmma(const float* __restrict__ u,
                      const float* __restrict__ x0,
                      const float* __restrict__ W1,
                      const float* __restrict__ b1,
                      const float* __restrict__ W2,
                      const float* __restrict__ b2,
                      const float* __restrict__ W3,
                      const float* __restrict__ b3,
                      const float* __restrict__ xmean,
                      const float* __restrict__ xstd,
                      const float* __restrict__ umean,
                      const float* __restrict__ ustd,
                      float* __restrict__ out)
{
  const int lane = threadIdx.x & 31;
  const int hl   = lane >> 4;     // half-wave index
  const int col  = lane & 15;     // batch column within the 16-wide tile
  const int n    = blockIdx.x * 16 + col;

  // ---- invariant scalars -------------------------------------------------
  const float xm0 = xmean[0], xm1 = xmean[1], xm2 = xmean[2];
  const float sx0 = xstd[0],  sx1 = xstd[1],  sx2 = xstd[2];
  const float ix0 = 1.0f / sx0, ix1 = 1.0f / sx1, ix2 = 1.0f / sx2;
  const float um  = umean[0], us = ustd[0];
  const float b30 = b3[0], b31 = b3[1];

  // ---- W2^T as A-fragments (f16), 4 output tiles x 2 K-halves ------------
  // f16 A 16x32, contiguous k order chosen here (matches B2): lane row m = lane&15; element e -> K =
  //   e + 16*(lane>=16), within a 32-wide K half.
  Frag aW2[4][2];
#pragma unroll
  for (int t = 0; t < 4; ++t) {
    const int m = 16 * t + col;
#pragma unroll
    for (int h = 0; h < 2; ++h) {
#pragma unroll
      for (int j = 0; j < 8; ++j) {
        const int e = 2 * j;
        const int K = 32 * h + e + 16 * hl;
        aW2[t][h].u[j] = pk2(W2[(size_t)K * 64 + m],
                             W2[(size_t)(K + 1) * 64 + m]);
      }
    }
  }

  // ---- W1^T (+ b1 folded as K=3 row) as A-fragments (f16) ----------------
  // Only K=0..3 are nonzero -> upper half-wave (K=8..15,24..31) is all zero.
  Frag aW1[4];
#pragma unroll
  for (int t = 0; t < 4; ++t) {
    const int m = 16 * t + col;
    const uint32_t p0 = pk2(W1[m], W1[64 + m]);
    const uint32_t p1 = pk2(W1[128 + m], b1[m]);
    aW1[t].u[0] = hl ? 0u : p0;
    aW1[t].u[1] = hl ? 0u : p1;
#pragma unroll
    for (int j = 2; j < 8; ++j) aW1[t].u[j] = 0u;
  }

  // ---- b2 as initial C fragments (C/D layout: dim = 16t + v + 8*hl) ------
  v8f c2[4];
#pragma unroll
  for (int t = 0; t < 4; ++t)
#pragma unroll
    for (int v = 0; v < 8; ++v)
      c2[t][v] = b2[16 * t + v + 8 * hl];

  // ---- per-lane W3 slice for the 32 hidden dims this lane owns -----------
  float w3a[4][8], w3b[4][8];
#pragma unroll
  for (int t = 0; t < 4; ++t)
#pragma unroll
    for (int v = 0; v < 8; ++v) {
      const int d = 16 * t + v + 8 * hl;
      w3a[t][v] = W3[2 * d];
      w3b[t][v] = W3[2 * d + 1];
    }

  // ---- state (replicated across both half-waves) -------------------------
  float X0v = x0[3 * n], X1v = x0[3 * n + 1], X2v = x0[3 * n + 2];
  const v8f zc = {};

#pragma unroll 1
  for (int ts = 0; ts < kT; ++ts) {
    // output is the state *entering* this step
    if (hl == 0) {
      const size_t o = ((size_t)n * kT + ts) * 3;
      out[o]     = X0v;
      out[o + 1] = X1v;
      out[o + 2] = X2v;
    }
    const float uv  = u[(size_t)n * kT + ts];
    const float Caf = uv * us + um;

    float k0 = 0.f, k1 = 0.f, k2 = 0.f;
    float a0 = 0.f, a1 = 0.f, a2 = 0.f;

#pragma unroll
    for (int s = 0; s < 4; ++s) {
      const float cs = (s == 0) ? 0.0f : ((s == 3) ? kDelta : 0.5f * kDelta);
      const float y0 = X0v + cs * k0;
      const float y1 = X1v + cs * k1;
      const float y2 = X2v + cs * k2;

      // ---- layer 1: D1 = (W1^T|b1) x [y0,y1,y2,1]^T ----------------------
      Frag B1;
      {
        const uint32_t p0 = pk2(y0, y1);
        const uint32_t p1 = pk2(y2, 1.0f);
        B1.u[0] = hl ? 0u : p0;
        B1.u[1] = hl ? 0u : p1;
#pragma unroll
        for (int j = 2; j < 8; ++j) B1.u[j] = 0u;
      }

      uint32_t hp[4][4];  // tanh(h1) packed half2, per tile, paired along v
#pragma unroll
      for (int t = 0; t < 4; ++t) {
        v8f d1 = __builtin_amdgcn_wmma_f32_16x16x32_f16(
            false, aW1[t].h, false, B1.h, (short)0, zc, false, false);
#pragma unroll
        for (int j = 0; j < 4; ++j)
          hp[t][j] = pk2(fast_tanh(d1[2 * j]), fast_tanh(d1[2 * j + 1]));
      }

      // ---- D-layout -> B-layout: half-wave exchange via ds_swizzle -------
      // B half h needs dims 32h+16*hl .. +15 = tile g = 2h+hl; lower 8
      // elements are hw=0 rows, upper 8 are hw=1 rows of that tile.
      Frag B2[2];
#pragma unroll
      for (int h = 0; h < 2; ++h) {
        const int g0 = 2 * h, g1 = 2 * h + 1;
#pragma unroll
        for (int j = 0; j < 4; ++j) {
          const uint32_t send = hl ? hp[g0][j] : hp[g1][j];
          const uint32_t recv = swapx16_u32(send);
          B2[h].u[j]     = hl ? recv      : hp[g0][j];
          B2[h].u[4 + j] = hl ? hp[g1][j] : recv;
        }
      }

      // ---- layer 2 (K=64 as two x32 WMMAs, C starts at b2) + layer 3 -----
      float r1p = 0.f, r2p = 0.f;
#pragma unroll
      for (int t = 0; t < 4; ++t) {
        v8f d2 = __builtin_amdgcn_wmma_f32_16x16x32_f16(
            false, aW2[t][0].h, false, B2[0].h, (short)0, c2[t], false, false);
        d2 = __builtin_amdgcn_wmma_f32_16x16x32_f16(
            false, aW2[t][1].h, false, B2[1].h, (short)0, d2, false, false);
#pragma unroll
        for (int v = 0; v < 8; ++v) {
          const float hv = fast_tanh(d2[v]);
          r1p = __builtin_fmaf(hv, w3a[t][v], r1p);
          r2p = __builtin_fmaf(hv, w3b[t][v], r2p);
        }
      }
      const float r1 = r1p + swapx16_f32(r1p) + b30;
      const float r2 = r2p + swapx16_f32(r2p) + b31;

      // ---- physics + normalization ---------------------------------------
      const float Ca = y0 * sx0 + xm0;
      const float Cb = y1 * sx1 + xm1;
      const float Cc = y2 * sx2 + xm2;
      k0 = (Caf - Ca) * kInvTau * ix0 - r1;
      k1 = -Cb * kInvTau * ix1 + (r1 * sx0 - 3.0f * r2 * sx2) * ix1;
      k2 = -Cc * kInvTau * ix2 + r2;

      const float w = (s == 0 || s == 3) ? (kDelta / 6.0f) : (kDelta / 3.0f);
      a0 += w * k0;
      a1 += w * k1;
      a2 += w * k2;
    }
    X0v += a0;
    X1v += a1;
    X2v += a2;
  }
}

extern "C" void kernel_launch(void* const* d_in, const int* in_sizes, int n_in,
                              void* d_out, int out_size, void* d_ws, size_t ws_size,
                              hipStream_t stream) {
  (void)in_sizes; (void)n_in; (void)d_ws; (void)ws_size; (void)out_size;
  const float* u     = (const float*)d_in[0];
  const float* x0    = (const float*)d_in[1];
  const float* W1    = (const float*)d_in[2];
  const float* b1    = (const float*)d_in[3];
  const float* W2    = (const float*)d_in[4];
  const float* b2    = (const float*)d_in[5];
  const float* W3    = (const float*)d_in[6];
  const float* b3    = (const float*)d_in[7];
  const float* xmean = (const float*)d_in[8];
  const float* xstd  = (const float*)d_in[9];
  const float* umean = (const float*)d_in[10];
  const float* ustd  = (const float*)d_in[11];
  float* out = (float*)d_out;

  // 2048 trajectories / 16 per wave = 128 independent single-wave workgroups.
  tworeac_rk4_wmma<<<dim3(kB / 16), dim3(32), 0, stream>>>(
      u, x0, W1, b1, W2, b2, W3, b3, xmean, xstd, umean, ustd, out);
}
